// DK_LSTM_Student_Pytorch_14035953123496
// MI455X (gfx1250) — matrix operand order repaired and checked by reading, not compiled, not run on hardware
//
#include <hip/hip_runtime.h>
#include <hip/hip_bf16.h>

typedef _Float16 v16h __attribute__((ext_vector_type(16)));
typedef _Float16 v8h  __attribute__((ext_vector_type(8)));
typedef float    v8f  __attribute__((ext_vector_type(8)));

#define B_TOT   64
#define T_LEN   16384
#define H1      64
#define H2      8
#define NG1     256     // 4*H1
#define NG2     32      // 4*H2
#define MB      16      // batch rows per workgroup
#define XCHUNK  128     // x timesteps staged in LDS per refill

// ---- fast activations: hardware TRANS ops, no IEEE division ----
__device__ __forceinline__ float fast_rcp(float x) {
    return __builtin_amdgcn_rcpf(x);            // v_rcp_f32
}
__device__ __forceinline__ float fast_sigmoid(float x) {
    return fast_rcp(1.0f + __expf(-x));         // v_exp_f32 + v_rcp_f32
}
__device__ __forceinline__ float fast_tanh(float x) {
#if __has_builtin(__builtin_amdgcn_tanhf)
    return __builtin_amdgcn_tanhf(x);           // v_tanh_f32 (CDNA5 TRANS op)
#else
    float ax = fabsf(x);
    float e  = __expf(-2.0f * ax);              // e in (0,1]
    float t  = fmaf(-2.0f * e, fast_rcp(1.0f + e), 1.0f);
    return copysignf(t, x);
#endif
}

__global__ __launch_bounds__(256)
void DK_LSTM_Student_Pytorch_14035953123496_kernel(
        const float* __restrict__ x,      // [B, T, 1]
        const float* __restrict__ W_ih1,  // [256, 1]
        const float* __restrict__ W_hh1,  // [256, 64]
        const float* __restrict__ b1,     // [256]
        const float* __restrict__ W_ih2,  // [32, 64]
        const float* __restrict__ W_hh2,  // [32, 8]
        const float* __restrict__ b2,     // [32]
        const float* __restrict__ W_out,  // [1, 8]
        const float* __restrict__ b_out,  // [1]
        float* __restrict__ out)          // y | h1 | c1 | h2 | c2
{
    __shared__ float                  s_g1[MB][NG1];   // activated layer-1 gates (16KB)
    __shared__ float                  s_g2[MB][NG2];   // layer-2 gates (2KB)
    __shared__ __align__(16) _Float16 s_h1h[MB][H1];   // f16 h1 (WMMA A source, 2KB)
    __shared__ float                  s_h1f[MB][H1];   // f32 h1 (final output, 4KB)
    __shared__ float                  s_h2[MB][H2];    // f32 h2 (512B)
    __shared__ float                  s_xs[XCHUNK][MB];// staged x (8KB)
    __shared__ float                  s_wih1[NG1];
    __shared__ float                  s_b1[NG1];
    __shared__ float                  s_b2[NG2];
    __shared__ float                  s_whh2[NG2][H2];
    __shared__ float                  s_wout[H2];
    __shared__ float                  s_bout;

    const int tid  = threadIdx.x;
    const int lane = tid & 31;          // wave32
    const int wid  = tid >> 5;          // 0..7
    const int swid = __builtin_amdgcn_readfirstlane(wid);  // scalar wave id
    const int b0   = blockIdx.x * MB;   // batch base for this WG

    // ---------------- setup: constants to LDS, states to zero ----------------
    for (int i = tid; i < NG1; i += 256) { s_wih1[i] = W_ih1[i]; s_b1[i] = b1[i]; }
    if (tid < NG2) s_b2[tid] = b2[tid];
    for (int i = tid; i < NG2 * H2; i += 256) s_whh2[i >> 3][i & 7] = W_hh2[i];
    if (tid < H2) s_wout[tid] = W_out[tid];
    if (tid == 0) s_bout = b_out[0];
    for (int i = tid; i < MB * H1; i += 256) {
        s_h1h[i >> 6][i & 63] = (_Float16)0.0f;
        s_h1f[i >> 6][i & 63] = 0.0f;
    }
    for (int i = tid; i < MB * H2; i += 256) s_h2[i >> 3][i & 7] = 0.0f;

    // ---------------- resident B fragments (f16) in VGPRs ----------------
    // Wave w owns gate columns [32w, 32w+32): entirely one gate class (i/f/g/o).
    const int col   = lane & 15;              // N within tile / M row for A
    const int khalf = (lane >> 4) & 1;        // which K half this lane holds
    v16h B1[2][2];                            // [k-chunk][local n-tile]
    for (int kc = 0; kc < 2; ++kc)
        for (int ntl = 0; ntl < 2; ++ntl) {
            int g = wid * 32 + ntl * 16 + col;
            const float* wp = W_hh1 + g * H1 + kc * 32 + khalf * 8;
            v16h tv;
            #pragma unroll
            for (int e = 0; e < 16; ++e) tv[e] = (_Float16)wp[e + (e >> 3) * 8];
            B1[kc][ntl] = tv;
        }
    v16h B2f[2];                              // layer-2 (waves 0,1 own n-tile = wid)
    if (swid < 2) {
        for (int kc = 0; kc < 2; ++kc) {
            int g = wid * 16 + col;
            const float* wp = W_ih2 + g * H1 + kc * 32 + khalf * 8;
            v16h tv;
            #pragma unroll
            for (int e = 0; e < 16; ++e) tv[e] = (_Float16)wp[e + (e >> 3) * 8];
            B2f[kc] = tv;
        }
    }

    // Cell states in registers
    float c1r[4];                             // pair p=tid+256k -> b=p>>6, h=p&63
    #pragma unroll
    for (int k = 0; k < 4; ++k) c1r[k] = 0.0f;
    float c2r = 0.0f;                         // threads 0..127 -> b=tid>>3, u=tid&7

    const int  mrow     = lane & 15;          // A-matrix row for this lane
    const int  mhi      = (lane >> 4) * 8;    // C/D row offset (0 or 8)
    const bool actTanh1 = (swid == 4 || swid == 5);   // scalar: 'g' gate waves
    // layer-2 combine: per-thread gate class (loop-invariant)
    const int  g2c      = tid & 31;
    const bool g2isG    = (g2c >= 16 && g2c < 24);

    __syncthreads();

    // ================================ time loop ================================
    #pragma unroll 1
    for (int t = 0; t < T_LEN; ++t) {
        const int tc = t & (XCHUNK - 1);
        if (tc == 0) {                         // refill staged x (coalesced)
            __syncthreads();
            for (int e = tid; e < MB * XCHUNK; e += 256) {
                int i = e / XCHUNK, j = e % XCHUNK;
                s_xs[j][i] = x[(size_t)(b0 + i) * T_LEN + t + j];
            }
            __syncthreads();
        }

        // ---- layer 1: G1 = x*W_ih1 + b1 + H1_prev @ W_hh1^T  (WMMA) ----
        v16h A1[2];
        #pragma unroll
        for (int kc = 0; kc < 2; ++kc) {
            const _Float16* p = &s_h1h[mrow][kc * 32 + khalf * 8];
            v16h a;
            *((v8h*)&a)       = *((const v8h*)p);        // ds_load_b128
            *(((v8h*)&a) + 1) = *((const v8h*)(p + 16)); // ds_load_b128
            A1[kc] = a;
        }
        float xv[8];
        #pragma unroll
        for (int j = 0; j < 8; ++j) xv[j] = s_xs[tc][mhi + j];

        #pragma unroll
        for (int ntl = 0; ntl < 2; ++ntl) {
            int g = wid * 32 + ntl * 16 + col;
            float wg = s_wih1[g], bg = s_b1[g];
            v8f acc;
            #pragma unroll
            for (int j = 0; j < 8; ++j) acc[j] = fmaf(xv[j], wg, bg);
            acc = __builtin_amdgcn_wmma_f32_16x16x32_f16(
                      false, A1[0], false, B1[0][ntl], (short)0, acc, false, false);
            acc = __builtin_amdgcn_wmma_f32_16x16x32_f16(
                      false, A1[1], false, B1[1][ntl], (short)0, acc, false, false);
            // gate nonlinearity in-register; scalar (wave-uniform) branch
            if (actTanh1) {
                #pragma unroll
                for (int j = 0; j < 8; ++j) s_g1[mhi + j][g] = fast_tanh(acc[j]);
            } else {
                #pragma unroll
                for (int j = 0; j < 8; ++j) s_g1[mhi + j][g] = fast_sigmoid(acc[j]);
            }
        }
        __syncthreads();

        // ---- layer-1 cell update: 1024 (b,h) pairs, 4 per thread ----
        #pragma unroll
        for (int k = 0; k < 4; ++k) {
            int p = tid + 256 * k;
            int bb = p >> 6, hh = p & 63;
            float ia = s_g1[bb][hh];
            float fa = s_g1[bb][64 + hh];
            float ga = s_g1[bb][128 + hh];
            float oa = s_g1[bb][192 + hh];
            float c  = fmaf(fa, c1r[k], ia * ga);
            c1r[k] = c;
            float h  = oa * fast_tanh(c);
            s_h1h[bb][hh] = (_Float16)h;
            s_h1f[bb][hh] = h;
        }
        __syncthreads();

        // ---- layer 2: h1-part via WMMA on waves 0,1 (scalar branch) ----
        if (swid < 2) {
            v16h A2[2];
            #pragma unroll
            for (int kc = 0; kc < 2; ++kc) {
                const _Float16* p = &s_h1h[mrow][kc * 32 + khalf * 8];
                v16h a;
                *((v8h*)&a)       = *((const v8h*)p);
                *(((v8h*)&a) + 1) = *((const v8h*)(p + 16));
                A2[kc] = a;
            }
            int g = wid * 16 + col;
            float bg = s_b2[g];
            v8f acc;
            #pragma unroll
            for (int j = 0; j < 8; ++j) acc[j] = bg;
            acc = __builtin_amdgcn_wmma_f32_16x16x32_f16(
                      false, A2[0], false, B2f[0], (short)0, acc, false, false);
            acc = __builtin_amdgcn_wmma_f32_16x16x32_f16(
                      false, A2[1], false, B2f[1], (short)0, acc, false, false);
            #pragma unroll
            for (int j = 0; j < 8; ++j) s_g2[mhi + j][g] = acc[j];
        }
        __syncthreads();

        // ---- add h2 @ W_hh2^T (K=8, VALU) and activate: 512 entries ----
        // branchless gate class: tanh(x) = 2*sigmoid(2x) - 1
        #pragma unroll
        for (int k = 0; k < 2; ++k) {
            int e  = tid + 256 * k;
            int bb = e >> 5;
            float v = s_g2[bb][g2c];
            #pragma unroll
            for (int u = 0; u < H2; ++u) v = fmaf(s_h2[bb][u], s_whh2[g2c][u], v);
            float uu = g2isG ? (2.0f * v) : v;        // v_cndmask / dual mul
            float s  = fast_sigmoid(uu);
            float r  = g2isG ? fmaf(2.0f, s, -1.0f) : s;
            s_g2[bb][g2c] = r;
        }
        __syncthreads();

        // ---- layer-2 cell update: 128 (b,u) pairs on threads 0..127 ----
        if (tid < 128) {
            int bb = tid >> 3, u = tid & 7;
            float ia = s_g2[bb][u];
            float fa = s_g2[bb][8 + u];
            float ga = s_g2[bb][16 + u];
            float oa = s_g2[bb][24 + u];
            float c  = fmaf(fa, c2r, ia * ga);
            c2r = c;
            s_h2[bb][u] = oa * fast_tanh(c);
        }
        __syncthreads();

        // ---- output head + residual: one thread per batch row ----
        if (tid < MB) {
            float acc = s_bout + s_xs[tc][tid];
            #pragma unroll
            for (int u = 0; u < H2; ++u) acc = fmaf(s_h2[tid][u], s_wout[u], acc);
            out[(size_t)(b0 + tid) * T_LEN + t] = acc;
        }
    }

    // ---------------- final states ----------------
    __syncthreads();
    const size_t O_h1 = (size_t)B_TOT * T_LEN;
    const size_t O_c1 = O_h1 + (size_t)B_TOT * H1;
    const size_t O_h2 = O_c1 + (size_t)B_TOT * H1;
    const size_t O_c2 = O_h2 + (size_t)B_TOT * H2;
    #pragma unroll
    for (int k = 0; k < 4; ++k) {
        int p = tid + 256 * k;
        int bb = p >> 6, hh = p & 63;
        out[O_h1 + (size_t)(b0 + bb) * H1 + hh] = s_h1f[bb][hh];
        out[O_c1 + (size_t)(b0 + bb) * H1 + hh] = c1r[k];
    }
    if (tid < 128) {
        int bb = tid >> 3, u = tid & 7;
        out[O_h2 + (size_t)(b0 + bb) * H2 + u] = s_h2[bb][u];
        out[O_c2 + (size_t)(b0 + bb) * H2 + u] = c2r;
    }
}

extern "C" void kernel_launch(void* const* d_in, const int* in_sizes, int n_in,
                              void* d_out, int out_size, void* d_ws, size_t ws_size,
                              hipStream_t stream) {
    const float* x     = (const float*)d_in[0];
    const float* W_ih1 = (const float*)d_in[1];
    const float* W_hh1 = (const float*)d_in[2];
    const float* b1    = (const float*)d_in[3];
    const float* W_ih2 = (const float*)d_in[4];
    const float* W_hh2 = (const float*)d_in[5];
    const float* b2    = (const float*)d_in[6];
    const float* W_out = (const float*)d_in[7];
    const float* b_out = (const float*)d_in[8];
    float* out = (float*)d_out;
    (void)in_sizes; (void)n_in; (void)out_size; (void)d_ws; (void)ws_size;

    DK_LSTM_Student_Pytorch_14035953123496_kernel<<<B_TOT / MB, 256, 0, stream>>>(
        x, W_ih1, W_hh1, b1, W_ih2, W_hh2, b2, W_out, b_out, out);
}
